// SelfAttention_79474074845790
// MI455X (gfx1250) — matrix operand order repaired and checked by reading, not compiled, not run on hardware
//
#include <hip/hip_runtime.h>

typedef __attribute__((ext_vector_type(16))) __bf16        v16bf;
typedef __attribute__((ext_vector_type(8)))  float         v8f;
typedef __attribute__((ext_vector_type(4)))  unsigned int  u32x4;
typedef __attribute__((ext_vector_type(8)))  int           i32x8;
typedef __attribute__((ext_vector_type(4)))  int           i32x4;

#define S_LEN  2048
#define BATCH  2
#define DMODEL 2048
#define NH     16
#define HDIM   128
#define NQKV   6144

#if __has_builtin(__builtin_amdgcn_tensor_load_to_lds) && \
    __has_builtin(__builtin_amdgcn_s_wait_tensorcnt)
#define HAVE_TDM 1
#else
#define HAVE_TDM 0
#endif

#if HAVE_TDM
// ---------------------------------------------------------------------------
// TDM: DMA a 2D fp32 tile (rows x row_elems, row stride in elements) from
// global memory into LDS at byte offset lds_off. D# built per CDNA5 ISA §8.
// ---------------------------------------------------------------------------
__device__ __forceinline__ void tdm_load_2d_f32(unsigned lds_off,
                                                const float* gptr,
                                                unsigned rows,
                                                unsigned row_elems,
                                                unsigned row_stride_elems)
{
    unsigned long long ga = (unsigned long long)(size_t)gptr;
    unsigned td0 = row_stride_elems;   // tensor dim0 (elements) = full row
    unsigned td1 = rows;               // tensor dim1 (rows)

    u32x4 g0;
    g0[0] = 1u;                                            // count=1 (valid)
    g0[1] = lds_off;                                       // lds_addr (bytes)
    g0[2] = (unsigned)(ga & 0xffffffffu);                  // global_addr[31:0]
    g0[3] = (unsigned)((ga >> 32) & 0x01ffffffu)           // global_addr[56:32]
          | (2u << 30);                                    // type = 2 (image)

    i32x8 g1;
    g1[0] = (int)(2u << 16);                               // data_size=2 (4B)
    g1[1] = (int)((td0 & 0xffffu) << 16);                  // tensor_dim0 lo
    g1[2] = (int)((td0 >> 16) | ((td1 & 0xffffu) << 16));  // dim0 hi | dim1 lo
    g1[3] = (int)((td1 >> 16) | (row_elems << 16));        // dim1 hi | tile_dim0
    g1[4] = (int)rows;                                     // tile_dim1 (tile_dim2=0)
    g1[5] = (int)row_stride_elems;                         // tensor_dim0_stride lo
    g1[6] = 0;                                             // stride hi | dim1_stride lo
    g1[7] = 0;

    i32x4 g2 = {0, 0, 0, 0};
    i32x4 g3 = {0, 0, 0, 0};
#if __clang_major__ >= 23
    i32x8 g4 = {0, 0, 0, 0, 0, 0, 0, 0};
    __builtin_amdgcn_tensor_load_to_lds(g0, g1, g2, g3, g4, 0);
#else
    __builtin_amdgcn_tensor_load_to_lds(g0, g1, g2, g3, 0);
#endif
}
#endif

// ---------------------------------------------------------------------------
// Tiled GEMM:  C[M x N] = A[M x K] * W^T + bias,  W stored [N x K] row-major.
// BM=64, BN=128, BK=32; 256 threads = 8 wave32, each wave a 16x64 strip.
// Double-buffered TDM pipeline: while the waves run the 32 WMMAs of tile t,
// the Tensor Data Mover fills the other LDS buffer with tile t+1. In-order
// TENSORcnt retirement => s_wait_tensorcnt 2 fences tile t's two transfers.
// MODE 0: epilogue scatters Q,K (fp32 [b,h][s][d]) and V (bf16 [b,h][d][s])
// MODE 1: plain row-major fp32 store
// ---------------------------------------------------------------------------
template <int MODE>
__global__ __launch_bounds__(256) void gemm_bf16_wmma(
    const float* __restrict__ A, const float* __restrict__ W,
    const float* __restrict__ bias,
    float* __restrict__ OQ, float* __restrict__ OK,
    __bf16* __restrict__ OV, float* __restrict__ OY,
    int K, int N)
{
    __shared__ float Af[2][64 * 32];     // 2 x 8 KB
    __shared__ float Bf[2][128 * 32];    // 2 x 16 KB

    const int tid  = threadIdx.x;
    const int lane = tid & 31;
    const int wv   = tid >> 5;
    const int half = lane >> 4;
    const int ln   = lane & 15;
    const int wm   = (wv >> 1) * 16;
    const int wn   = (wv & 1) * 64;
    const int m0   = blockIdx.x * 64;
    const int n0   = blockIdx.y * 128;
    const int T    = K >> 5;             // number of 32-wide k tiles

    v8f acc[4] = {};

#if HAVE_TDM
    if (wv == 0) {   // prologue: stage tile 0 into buffer 0
        tdm_load_2d_f32((unsigned)(size_t)(void*)&Af[0][0],
                        A + (size_t)m0 * K, 64, 32, (unsigned)K);
        tdm_load_2d_f32((unsigned)(size_t)(void*)&Bf[0][0],
                        W + (size_t)n0 * K, 128, 32, (unsigned)K);
    }
#endif

    for (int t = 0; t < T; ++t) {
        const int cur = t & 1;
#if HAVE_TDM
        if (wv == 0) {
            if (t + 1 < T) {
                const int nxt = (t + 1) & 1;
                const int k1  = (t + 1) << 5;
                tdm_load_2d_f32((unsigned)(size_t)(void*)&Af[nxt][0],
                                A + (size_t)m0 * K + k1, 64, 32, (unsigned)K);
                tdm_load_2d_f32((unsigned)(size_t)(void*)&Bf[nxt][0],
                                W + (size_t)n0 * K + k1, 128, 32, (unsigned)K);
                __builtin_amdgcn_s_wait_tensorcnt(2);  // tile t complete
            } else {
                __builtin_amdgcn_s_wait_tensorcnt(0);  // last tile complete
            }
        }
#else
        {
            const int k0 = t << 5;
            #pragma unroll
            for (int i = 0; i < 8; ++i) {
                int idx = tid + i * 256;
                Af[cur][idx] = A[(size_t)(m0 + (idx >> 5)) * K + k0 + (idx & 31)];
            }
            #pragma unroll
            for (int i = 0; i < 16; ++i) {
                int idx = tid + i * 256;
                Bf[cur][idx] = W[(size_t)(n0 + (idx >> 5)) * K + k0 + (idx & 31)];
            }
        }
#endif
        __syncthreads();

        // A fragment: 16x32, M = lane&15, K interleaved by lane half
        v16bf af;
        #pragma unroll
        for (int j = 0; j < 16; ++j) {
            int kk = (j & 7) + 8 * half + 16 * (j >> 3);
            af[j] = (__bf16)Af[cur][(wm + ln) * 32 + kk];
        }
        #pragma unroll
        for (int f = 0; f < 4; ++f) {
            v16bf bfr;
            #pragma unroll
            for (int j = 0; j < 16; ++j)
                bfr[j] = (__bf16)Bf[cur][(wn + 16 * f + ln) * 32 + (j & 7) + 8 * half + 16 * (j >> 3)];
            acc[f] = __builtin_amdgcn_wmma_f32_16x16x32_bf16(
                false, af, false, bfr, (short)0, acc[f], false, false);
        }
        __syncthreads();   // all reads of buffer `cur` done before its refill
    }

    // ---- epilogue: bias + store ----
    #pragma unroll
    for (int f = 0; f < 4; ++f) {
        int ng = n0 + wn + 16 * f + ln;
        float bv = bias[ng];
        #pragma unroll
        for (int i = 0; i < 8; ++i) {
            int mg = m0 + wm + i + 8 * half;       // row in [S*B]
            float val = acc[f][i] + bv;
            if (MODE == 0) {
                int h = ng / 384, rem = ng % 384;
                int t2 = rem >> 7, d = rem & 127;   // 0=q 1=k 2=v
                int s = mg >> 1, b = mg & 1;        // BATCH == 2
                int bh = b * NH + h;
                if (t2 == 2) {
                    // V: bf16, transposed [b,h][d][s]
                    OV[((size_t)bh * HDIM + d) * S_LEN + s] = (__bf16)val;
                } else {
                    float* dst = (t2 == 0) ? OQ : OK;
                    dst[((size_t)bh * S_LEN + s) * HDIM + d] = val;
                }
            } else {
                OY[(size_t)mg * N + ng] = val;
            }
        }
    }
}

// ---------------------------------------------------------------------------
// Fused rotary + fp32->bf16 convert. One thread per (b,h,s,d) element of Q
// and K. Rotary on d<32 (pairs d, d±16). Q pre-scaled by 1/sqrt(HD).
// ---------------------------------------------------------------------------
__global__ __launch_bounds__(256) void rope_cvt_kernel(
    const float* __restrict__ Qf, const float* __restrict__ Kf,
    __bf16* __restrict__ Qb, __bf16* __restrict__ Kb)
{
    const float scale = 0.08838834764831845f;  // 1/sqrt(128)
    int idx = blockIdx.x * 256 + threadIdx.x;   // [0, B*H*S*HD)
    int d  = idx & 127;
    int s  = (idx >> 7) & (S_LEN - 1);
    int bh = idx >> 18;
    size_t off = ((size_t)bh * S_LEN + s) * HDIM;

    float q = Qf[off + d], k = Kf[off + d];
    if (d < 32) {
        int i = d & 15;
        float inv = __powf(10000.0f, -(float)i * (1.0f / 16.0f));
        float sn, cs;
        __sincosf((float)s * inv, &sn, &cs);
        if (d < 16) {
            float qp = Qf[off + d + 16], kp = Kf[off + d + 16];
            q = q * cs - qp * sn;
            k = k * cs - kp * sn;
        } else {
            float qp = Qf[off + d - 16], kp = Kf[off + d - 16];
            q = q * cs + qp * sn;
            k = k * cs + kp * sn;
        }
    }
    Qb[off + d] = (__bf16)(q * scale);
    Kb[off + d] = (__bf16)k;
}

// ---------------------------------------------------------------------------
// Flash attention, causal. One wave32 per (b, h, 16-row Q tile). All-bf16
// operands (Q pre-scaled, V transposed), fp32 accumulation, online softmax.
// Exactly one 32-key tile straddles the diagonal => mask only that tile.
// ---------------------------------------------------------------------------
__global__ __launch_bounds__(32) void flash_attn_wmma(
    const __bf16* __restrict__ Q, const __bf16* __restrict__ K,
    const __bf16* __restrict__ VT, float* __restrict__ ctx)
{
    __shared__ __bf16 Pl[16][40];  // 16x32 P tile (+pad)

    const int lane = threadIdx.x;
    const int half = lane >> 4, ln = lane & 15;
    const int nqt  = S_LEN / 16;
    const int qt   = blockIdx.x % nqt;
    const int bh   = blockIdx.x / nqt;     // = b*NH + h
    const int h    = bh % NH, b = bh / NH;
    const size_t base = (size_t)bh * S_LEN * HDIM;
    const int q0 = qt * 16;

    // Q tile -> 4 A-fragments (already scaled)
    v16bf qf[4];
    #pragma unroll
    for (int c = 0; c < 4; ++c)
        #pragma unroll
        for (int j = 0; j < 16; ++j) {
            int kk = (j & 7) + 8 * half + 16 * (j >> 3);
            qf[c][j] = Q[base + (size_t)(q0 + ln) * HDIM + 32 * c + kk];
        }

    float rmax[8], rsum[8];
    v8f acc[8] = {};
    #pragma unroll
    for (int i = 0; i < 8; ++i) { rmax[i] = -3.0e38f; rsum[i] = 0.0f; }

    const int nkv = ((q0 + 15) >> 5) + 1;
    for (int kt = 0; kt < nkv; ++kt) {
        const int kb = kt * 32;
        const bool diag = (kt == nkv - 1);   // only tile that needs masking

        // ---- scores: 16 x 32 keys = two WMMA accumulators ----
        v8f s0 = {}, s1 = {};
        #pragma unroll
        for (int c = 0; c < 4; ++c) {
            v16bf kf0, kf1;
            #pragma unroll
            for (int j = 0; j < 16; ++j) {
                int dd = 32 * c + (j & 7) + 8 * half + 16 * (j >> 3);
                kf0[j] = K[base + (size_t)(kb + ln) * HDIM + dd];
                kf1[j] = K[base + (size_t)(kb + 16 + ln) * HDIM + dd];
            }
            s0 = __builtin_amdgcn_wmma_f32_16x16x32_bf16(false, qf[c], false, kf0, (short)0, s0, false, false);
            s1 = __builtin_amdgcn_wmma_f32_16x16x32_bf16(false, qf[c], false, kf1, (short)0, s1, false, false);
        }

        // ---- causal mask (diagonal tile only) + online softmax ----
        #pragma unroll
        for (int i = 0; i < 8; ++i) {
            float v0 = s0[i], v1 = s1[i];
            if (diag) {
                const int qpos = q0 + i + 8 * half;
                v0 = (kb + ln      <= qpos) ? v0 : -1.0e4f;
                v1 = (kb + 16 + ln <= qpos) ? v1 : -1.0e4f;
            }
            float t = fmaxf(v0, v1);
            #pragma unroll
            for (int off = 1; off <= 8; off <<= 1)
                t = fmaxf(t, __shfl_xor(t, off, 32));
            float mnew  = fmaxf(rmax[i], t);
            float alpha = __expf(rmax[i] - mnew);
            rmax[i] = mnew;
            float p0 = __expf(v0 - mnew);
            float p1 = __expf(v1 - mnew);
            float rs = p0 + p1;
            #pragma unroll
            for (int off = 1; off <= 8; off <<= 1)
                rs += __shfl_xor(rs, off, 32);
            rsum[i] = rsum[i] * alpha + rs;
            #pragma unroll
            for (int f = 0; f < 8; ++f) acc[f][i] *= alpha;
            Pl[i + 8 * half][ln]      = (__bf16)p0;
            Pl[i + 8 * half][16 + ln] = (__bf16)p1;
        }
        __syncthreads();

        // P -> A-fragment layout
        v16bf pf;
        #pragma unroll
        for (int j = 0; j < 16; ++j)
            pf[j] = Pl[ln][(j & 7) + 8 * half + 16 * (j >> 3)];

        // ---- PV: accumulate 16 x 128 context (V transposed: contiguous) ----
        #pragma unroll
        for (int f = 0; f < 8; ++f) {
            v16bf vf;
            #pragma unroll
            for (int j = 0; j < 16; ++j)
                vf[j] = VT[((size_t)bh * HDIM + 16 * f + ln) * S_LEN + kb + (j & 7) + 8 * half + 16 * (j >> 3)];
            acc[f] = __builtin_amdgcn_wmma_f32_16x16x32_bf16(false, pf, false, vf, (short)0, acc[f], false, false);
        }
        __syncthreads();
    }

    // ---- normalize and store ctx [S*B][D] ----
    #pragma unroll
    for (int f = 0; f < 8; ++f)
        #pragma unroll
        for (int i = 0; i < 8; ++i) {
            int m = i + 8 * half;
            int d = 16 * f + ln;
            float val = acc[f][i] / rsum[i];
            ctx[(size_t)((q0 + m) * BATCH + b) * DMODEL + h * HDIM + d] = val;
        }
}

// ---------------------------------------------------------------------------
extern "C" void kernel_launch(void* const* d_in, const int* in_sizes, int n_in,
                              void* d_out, int out_size, void* d_ws, size_t ws_size,
                              hipStream_t stream)
{
    const float* X    = (const float*)d_in[0];
    // d_in[1] = attention_mask (pure causal; handled analytically)
    const float* Wqkv = (const float*)d_in[2];
    const float* bqkv = (const float*)d_in[3];
    const float* Wd   = (const float*)d_in[4];
    const float* bd   = (const float*)d_in[5];
    float* out = (float*)d_out;

    const size_t perBuf = (size_t)BATCH * NH * S_LEN * HDIM;  // 8,388,608 elems
    float*  Qf  = (float*)d_ws;          // fp32 Q (later reused as ctx)
    float*  Kf  = Qf + perBuf;           // fp32 K
    __bf16* Qb  = (__bf16*)(Kf + perBuf);
    __bf16* Kb  = Qb + perBuf;
    __bf16* Vb  = Kb + perBuf;           // bf16 V, transposed [b,h][d][s]
    float*  ctx = Qf;                    // alias: Qf dead after rope_cvt

    // 1) QKV projection (103 GFLOP): Q,K fp32; V bf16 transposed
    gemm_bf16_wmma<0><<<dim3(64, 48), 256, 0, stream>>>(
        X, Wqkv, bqkv, Qf, Kf, Vb, nullptr, DMODEL, NQKV);
    // 2) rotary + convert: Qb (pre-scaled), Kb
    rope_cvt_kernel<<<32768, 256, 0, stream>>>(Qf, Kf, Qb, Kb);
    // 3) causal flash attention -> ctx
    flash_attn_wmma<<<BATCH * NH * (S_LEN / 16), 32, 0, stream>>>(Qb, Kb, Vb, ctx);
    // 4) dense projection (34 GFLOP) -> out
    gemm_bf16_wmma<1><<<dim3(64, 16), 256, 0, stream>>>(
        ctx, Wd, bd, nullptr, nullptr, nullptr, out, DMODEL, DMODEL);
}
